// NGNN_69028714381397
// MI455X (gfx1250) — matrix operand order repaired and checked by reading, not compiled, not run on hardware
//
#include <hip/hip_runtime.h>
#include <hip/hip_bf16.h>

// ---------------------------------------------------------------------------
// NGNN pipeline for MI455X (gfx1250, wave32, WMMA).
// GEMMs: v_wmma_f32_16x16x32_bf16, one wave = 16 rows x (4..6 x 16) cols so
// independent accumulator chains hide WMMA->WMMA hazards and A fragments are
// reused across all N-tiles.  Scatter-adds: global_atomic_add_f32.
// ---------------------------------------------------------------------------

typedef __bf16 bf16;
typedef __attribute__((ext_vector_type(16))) __bf16 v16bf;
typedef __attribute__((ext_vector_type(8)))  float  v8f;

static constexpr int NN = 500000;   // nodes
static constexpr int EE = 1250000;  // edges
static constexpr int SS = 50000;    // subgraphs
static constexpr int GG = 64;       // graphs
static constexpr int DD = 64;       // hidden
static constexpr int NL = 5;        // layers

__device__ __forceinline__ bf16 bf16_bits(unsigned short u) {
  union { unsigned short s; bf16 h; } c; c.s = u; return c.h;
}

// A fragment (16x32, 16-bit): lane r holds row M=m0+r, dword v holds
// K = 2*(v&3) + 16*(v>>2) + 8*hi (+half).  Source stored as bf16.
__device__ __forceinline__ void load_a_bf16(const bf16* __restrict__ rowA, int hi, v16bf& a) {
#pragma unroll
  for (int t = 0; t < 8; ++t) {
    int K = 2 * (t & 3) + 16 * (t >> 2) + 8 * hi;          // even -> dword aligned
    unsigned int w = *(const unsigned int*)(rowA + K);
    a[2 * t]     = bf16_bits((unsigned short)(w & 0xffffu));
    a[2 * t + 1] = bf16_bits((unsigned short)(w >> 16));
  }
}

// Same fragment but source stored as f32 (convert in-register).
__device__ __forceinline__ void load_a_f32(const float* __restrict__ rowA, int hi, v16bf& a) {
#pragma unroll
  for (int t = 0; t < 8; ++t) {
    int K = 2 * (t & 3) + 16 * (t >> 2) + 8 * hi;
    float2 f = *(const float2*)(rowA + K);
    a[2 * t]     = (bf16)f.x;
    a[2 * t + 1] = (bf16)f.y;
  }
}

// B fragment (32x16, 16-bit): lane holds column n0 + (kk&15), element e is
// K = k0 + 8*hi + e (e<8) or k0 + 16 + 8*hi + (e-8) (e>=8); Wt[k][n], ld nout.
__device__ __forceinline__ void load_b(const bf16* __restrict__ wt, int kk, int nout, int n0, v16bf& b) {
  int k0 = kk & ~31, c = kk & 15, h = (kk >> 4) & 1;
#pragma unroll
  for (int e = 0; e < 16; ++e) {
    int K = k0 + ((e < 8) ? (8 * h + e) : (16 + 8 * h + (e - 8)));
    b[e] = wt[(size_t)K * nout + n0 + c];
  }
}

#define WMMA_BF16(acc, a, b)                                                     \
  acc = __builtin_amdgcn_wmma_f32_16x16x32_bf16(false, a, false, b, (short)0,    \
                                                acc, false, false)

// C/D tile stores: VGPR j -> row m0 + j + 8*hi, col n0 + r.
__device__ __forceinline__ void store_tile_bf16(bf16* __restrict__ out, int ld, int m0,
                                                int n0, int r, int hi, float bn, const v8f& acc) {
#pragma unroll
  for (int j = 0; j < 8; ++j)
    out[(size_t)(m0 + j + 8 * hi) * ld + n0 + r] = (bf16)(acc[j] + bn);
}
__device__ __forceinline__ void store_tile_f32(float* __restrict__ out, int ld, int m0,
                                               int n0, int r, int hi, float bn, const v8f& acc) {
#pragma unroll
  for (int j = 0; j < 8; ++j)
    out[(size_t)(m0 + j + 8 * hi) * ld + n0 + r] = acc[j] + bn;
}
__device__ __forceinline__ void addto_tile_f32(float* __restrict__ out, int ld, int m0,
                                               int n0, int r, int hi, float bn, const v8f& acc) {
#pragma unroll
  for (int j = 0; j < 8; ++j) {
    size_t o = (size_t)(m0 + j + 8 * hi) * ld + n0 + r;
    out[o] = out[o] + acc[j] + bn;     // exclusive tile: plain RMW
  }
}

// ---------------------------------------------------------------------------
// Weight preparation
// ---------------------------------------------------------------------------
__global__ __launch_bounds__(256) void convert_copy_kernel(const float* __restrict__ src,
                                                           bf16* __restrict__ dst, int n) {
  int i = blockIdx.x * 256 + threadIdx.x;
  if (i < n) dst[i] = (bf16)src[i];
}

// src f32 [R][C] -> dst bf16 [C][R]
__global__ __launch_bounds__(256) void convert_transpose_kernel(const float* __restrict__ src,
                                                                bf16* __restrict__ dst,
                                                                int R, int C) {
  int i = blockIdx.x * 256 + threadIdx.x;
  if (i < R * C) {
    int r = i / C, c = i % C;
    dst[(size_t)c * R + r] = (bf16)src[i];
  }
}

__global__ __launch_bounds__(256) void zero_kernel(float* __restrict__ p, int n) {
  int i = blockIdx.x * 256 + threadIdx.x;
  if (i < n) p[i] = 0.0f;
}

// x0 = z_tables[0][z]  (bf16 gather-copy)
__global__ __launch_bounds__(256) void embed0_kernel(const int* __restrict__ z,
                                                     const bf16* __restrict__ tab0,
                                                     bf16* __restrict__ x) {
  int i = blockIdx.x * 256 + threadIdx.x;
  if (i < NN * DD) {
    int node = i >> 6, c = i & 63;
    x[i] = tab0[(size_t)z[node] * DD + c];
  }
}

// ---------------------------------------------------------------------------
// GEMM kernels.  One wave = 16 rows x full output width (64) or half (96).
// All guards are wave-uniform and precede every WMMA (EXEC all-ones).
// ---------------------------------------------------------------------------

// xout = concat(x, emb(z)) @ tfW.T + tfb    (K = 128, Nout = 64, 4 acc chains)
__global__ __launch_bounds__(256) void tf_gemm_kernel(const bf16* __restrict__ x,
                                                      const int* __restrict__ z,
                                                      const bf16* __restrict__ tab,
                                                      const bf16* __restrict__ wt,   // [128][64]
                                                      const float* __restrict__ bias,
                                                      bf16* __restrict__ xout) {
  int wave = (blockIdx.x * blockDim.x + threadIdx.x) >> 5;
  if (wave >= NN / 16) return;                       // wave-uniform
  int lane = threadIdx.x & 31, r = lane & 15, hi = lane >> 4;
  int m0 = wave * 16;
  int row = m0 + r;
  const bf16* xrow = x + (size_t)row * DD;
  const bf16* trow = tab + (size_t)z[row] * DD;
  v8f acc[4] = {};
#pragma unroll
  for (int s = 0; s < 4; ++s) {
    v16bf a;
    const bf16* src = (s < 2) ? (xrow + 32 * s) : (trow + 32 * (s - 2));
    load_a_bf16(src, hi, a);
    int kk = 32 * s + r + 16 * hi;
#pragma unroll
    for (int t = 0; t < 4; ++t) {
      v16bf b;
      load_b(wt, kk, DD, 16 * t, b);
      WMMA_BF16(acc[t], a, b);
    }
  }
#pragma unroll
  for (int t = 0; t < 4; ++t)
    store_tile_bf16(xout, DD, m0, 16 * t, r, hi, bias[16 * t + r], acc[t]);
}

// m = x @ ggcW         (K = 64, Nout = 64, f32 out, no bias, 4 acc chains)
__global__ __launch_bounds__(256) void ggc_gemm_kernel(const bf16* __restrict__ x,
                                                       const bf16* __restrict__ wt, // [64][64]
                                                       float* __restrict__ mbuf) {
  int wave = (blockIdx.x * blockDim.x + threadIdx.x) >> 5;
  if (wave >= NN / 16) return;
  int lane = threadIdx.x & 31, r = lane & 15, hi = lane >> 4;
  int m0 = wave * 16;
  const bf16* xrow = x + (size_t)(m0 + r) * DD;
  v8f acc[4] = {};
#pragma unroll
  for (int s = 0; s < 2; ++s) {
    v16bf a;
    load_a_bf16(xrow + 32 * s, hi, a);
    int kk = 32 * s + r + 16 * hi;
#pragma unroll
    for (int t = 0; t < 4; ++t) {
      v16bf b;
      load_b(wt, kk, DD, 16 * t, b);
      WMMA_BF16(acc[t], a, b);
    }
  }
#pragma unroll
  for (int t = 0; t < 4; ++t)
    store_tile_f32(mbuf, DD, m0, 16 * t, r, hi, 0.0f, acc[t]);
}

// gi = agg @ wih.T + bih   (K = 64, Nout = 192; A from f32; 6 acc chains,
// two waves per row block cover cols [0,96) and [96,192))
__global__ __launch_bounds__(256) void gi_gemm_kernel(const float* __restrict__ agg,
                                                      const bf16* __restrict__ wt, // [64][192]
                                                      const float* __restrict__ bias,
                                                      float* __restrict__ gi) {
  int wave = (blockIdx.x * blockDim.x + threadIdx.x) >> 5;
  if (wave >= (NN / 16) * 2) return;
  int lane = threadIdx.x & 31, r = lane & 15, hi = lane >> 4;
  int m0 = (wave >> 1) * 16, nbase = (wave & 1) * 96;
  const float* arow = agg + (size_t)(m0 + r) * DD;
  v8f acc[6] = {};
#pragma unroll
  for (int s = 0; s < 2; ++s) {
    v16bf a;
    load_a_f32(arow + 32 * s, hi, a);
    int kk = 32 * s + r + 16 * hi;
#pragma unroll
    for (int t = 0; t < 6; ++t) {
      v16bf b;
      load_b(wt, kk, 192, nbase + 16 * t, b);
      WMMA_BF16(acc[t], a, b);
    }
  }
#pragma unroll
  for (int t = 0; t < 6; ++t) {
    int n0 = nbase + 16 * t;
    store_tile_f32(gi, 192, m0, n0, r, hi, bias[n0 + r], acc[t]);
  }
}

// gh = x @ whh.T + bhh.  Cols 0..127 accumulate into gi (ir+hr, iz+hz);
// cols 128..191 (hn) go to the separate hn buffer (reuses m).
__global__ __launch_bounds__(256) void gh_gemm_kernel(const bf16* __restrict__ x,
                                                      const bf16* __restrict__ wt, // [64][192]
                                                      const float* __restrict__ bias,
                                                      float* __restrict__ gi,
                                                      float* __restrict__ hn) {
  int wave = (blockIdx.x * blockDim.x + threadIdx.x) >> 5;
  if (wave >= (NN / 16) * 2) return;
  int lane = threadIdx.x & 31, r = lane & 15, hi = lane >> 4;
  int m0 = (wave >> 1) * 16, nbase = (wave & 1) * 96;
  const bf16* xrow = x + (size_t)(m0 + r) * DD;
  v8f acc[6] = {};
#pragma unroll
  for (int s = 0; s < 2; ++s) {
    v16bf a;
    load_a_bf16(xrow + 32 * s, hi, a);
    int kk = 32 * s + r + 16 * hi;
#pragma unroll
    for (int t = 0; t < 6; ++t) {
      v16bf b;
      load_b(wt, kk, 192, nbase + 16 * t, b);
      WMMA_BF16(acc[t], a, b);
    }
  }
#pragma unroll
  for (int t = 0; t < 6; ++t) {
    int n0 = nbase + 16 * t;
    float bn = bias[n0 + r];
    if (n0 < 128) {                      // wave-uniform (after all WMMAs)
      addto_tile_f32(gi, 192, m0, n0, r, hi, bn, acc[t]);
    } else {
      store_tile_f32(hn, DD, m0, n0 - 128, r, hi, bn, acc[t]);
    }
  }
}

// ---------------------------------------------------------------------------
// Scatter / elementwise / pooling / head
// ---------------------------------------------------------------------------

// agg[dst] += m[src] over edges; one wave per edge, 2 channels per lane.
__global__ __launch_bounds__(256) void scatter_edges_kernel(const int* __restrict__ esrc,
                                                            const int* __restrict__ edst,
                                                            const float* __restrict__ mbuf,
                                                            float* __restrict__ agg) {
  int t = blockIdx.x * 256 + threadIdx.x;
  int e = t >> 5, lane = t & 31;
  if (e >= EE) return;
  int s = esrc[e], d = edst[e];
  float2 v = *(const float2*)(mbuf + (size_t)s * DD + 2 * lane);
  atomicAdd(&agg[(size_t)d * DD + 2 * lane], v.x);
  atomicAdd(&agg[(size_t)d * DD + 2 * lane + 1], v.y);
}

// GRU elementwise update, in-place on bf16 x.
__global__ __launch_bounds__(256) void gru_kernel(bf16* __restrict__ x,
                                                  const float* __restrict__ gi,
                                                  const float* __restrict__ hn) {
  int idx = blockIdx.x * 256 + threadIdx.x;
  if (idx >= NN * DD) return;
  int node = idx >> 6, c = idx & 63;
  const float* grow = gi + (size_t)node * 192;
  float r = 1.0f / (1.0f + __expf(-grow[c]));
  float u = 1.0f / (1.0f + __expf(-grow[64 + c]));
  float n = tanhf(grow[128 + c] + r * hn[idx]);
  float xf = (float)x[idx];
  x[idx] = (bf16)((1.0f - u) * n + u * xf);
}

__global__ __launch_bounds__(256) void pool_nodes_kernel(const bf16* __restrict__ x,
                                                         const int* __restrict__ n2s,
                                                         float* __restrict__ poolS) {
  int t = blockIdx.x * 256 + threadIdx.x;
  int n = t >> 5, lane = t & 31;
  if (n >= NN) return;
  unsigned int w = *(const unsigned int*)(x + (size_t)n * DD + 2 * lane);
  float a = (float)bf16_bits((unsigned short)(w & 0xffffu));
  float b = (float)bf16_bits((unsigned short)(w >> 16));
  int sg = n2s[n];
  atomicAdd(&poolS[(size_t)sg * DD + 2 * lane], a);
  atomicAdd(&poolS[(size_t)sg * DD + 2 * lane + 1], b);
}

__global__ __launch_bounds__(256) void pool_sub_kernel(const float* __restrict__ poolS,
                                                       const int* __restrict__ s2g,
                                                       float* __restrict__ poolG) {
  int t = blockIdx.x * 256 + threadIdx.x;
  int s = t >> 5, lane = t & 31;
  if (s >= SS) return;
  float2 v = *(const float2*)(poolS + (size_t)s * DD + 2 * lane);
  int g = s2g[s];
  atomicAdd(&poolG[(size_t)g * DD + 2 * lane], v.x);
  atomicAdd(&poolG[(size_t)g * DD + 2 * lane + 1], v.y);
}

__global__ void head_kernel(const float* __restrict__ pg,
                            const float* __restrict__ w1, const float* __restrict__ b1,
                            const float* __restrict__ w2, const float* __restrict__ b2,
                            const float* __restrict__ w3, const float* __restrict__ b3,
                            float* __restrict__ out) {
  int g = threadIdx.x;
  if (g >= GG) return;
  float xg[64];
  for (int j = 0; j < 64; ++j) xg[j] = pg[g * 64 + j];
  float h1[32];
  for (int k = 0; k < 32; ++k) {
    float s = b1[k];
    for (int j = 0; j < 64; ++j) s += w1[k * 64 + j] * xg[j];
    h1[k] = (s > 0.0f) ? s : expm1f(s);  // ELU
  }
  float h2[16];
  for (int k = 0; k < 16; ++k) {
    float s = b2[k];
    for (int j = 0; j < 32; ++j) s += w2[k * 32 + j] * h1[j];
    h2[k] = (s > 0.0f) ? s : expm1f(s);
  }
  float o = b3[0];
  for (int j = 0; j < 16; ++j) o += w3[j] * h2[j];
  out[g] = o;
}

// ---------------------------------------------------------------------------
// Host launch
// ---------------------------------------------------------------------------
static inline int cc(long long n) { return (int)((n + 255) / 256); }

extern "C" void kernel_launch(void* const* d_in, const int* in_sizes, int n_in,
                              void* d_out, int out_size, void* d_ws, size_t ws_size,
                              hipStream_t stream) {
  (void)in_sizes; (void)n_in; (void)out_size; (void)ws_size;

  const int*   z        = (const int*)d_in[0];
  const int*   ei       = (const int*)d_in[1];   // [2][E]
  const int*   n2s      = (const int*)d_in[2];
  const int*   s2g      = (const int*)d_in[3];
  const float* z_tables = (const float*)d_in[4]; // [5][100][64]
  const float* tf_W     = (const float*)d_in[5]; // [4][64][128]
  const float* tf_b     = (const float*)d_in[6]; // [4][64]
  const float* ggc_W    = (const float*)d_in[7]; // [5][64][64]
  const float* wih      = (const float*)d_in[8]; // [5][192][64]
  const float* whh      = (const float*)d_in[9]; // [5][192][64]
  const float* bih      = (const float*)d_in[10];// [5][192]
  const float* bhh      = (const float*)d_in[11];// [5][192]
  const float* fc1W = (const float*)d_in[12]; const float* fc1b = (const float*)d_in[13];
  const float* fc2W = (const float*)d_in[14]; const float* fc2b = (const float*)d_in[15];
  const float* fc3W = (const float*)d_in[16]; const float* fc3b = (const float*)d_in[17];
  float* out = (float*)d_out;

  const int* esrc = ei;
  const int* edst = ei + EE;

  // ---- workspace carve-up (deterministic) ----
  char* ws = (char*)d_ws;
  size_t off = 0;
  auto alloc = [&](size_t bytes) -> void* {
    off = (off + 255) & ~(size_t)255;
    void* p = ws + off;
    off += bytes;
    return p;
  };
  bf16*  xA     = (bf16*)alloc((size_t)NN * DD * 2);
  bf16*  xB     = (bf16*)alloc((size_t)NN * DD * 2);
  float* mbuf   = (float*)alloc((size_t)NN * DD * 4);   // also reused as hn
  float* agg    = (float*)alloc((size_t)NN * DD * 4);
  float* gi     = (float*)alloc((size_t)NN * 192 * 4);
  float* poolS  = (float*)alloc((size_t)SS * DD * 4);
  float* poolG  = (float*)alloc((size_t)GG * DD * 4);
  bf16*  tab_bf = (bf16*)alloc((size_t)NL * 100 * DD * 2);
  bf16*  tfWt   = (bf16*)alloc((size_t)4 * 128 * DD * 2);
  bf16*  ggcWt  = (bf16*)alloc((size_t)NL * DD * DD * 2);
  bf16*  wihWt  = (bf16*)alloc((size_t)NL * DD * 192 * 2);
  bf16*  whhWt  = (bf16*)alloc((size_t)NL * DD * 192 * 2);

  // ---- weight prep (bf16, [K][N] layout) ----
  convert_copy_kernel<<<cc(NL * 100 * DD), 256, 0, stream>>>(z_tables, tab_bf, NL * 100 * DD);
  for (int l = 0; l < NL; ++l) {
    if (l > 0)
      convert_transpose_kernel<<<cc(64 * 128), 256, 0, stream>>>(
          tf_W + (size_t)(l - 1) * 64 * 128, tfWt + (size_t)(l - 1) * 128 * 64, 64, 128);
    convert_copy_kernel<<<cc(64 * 64), 256, 0, stream>>>(
        ggc_W + (size_t)l * 64 * 64, ggcWt + (size_t)l * 64 * 64, 64 * 64);
    convert_transpose_kernel<<<cc(192 * 64), 256, 0, stream>>>(
        wih + (size_t)l * 192 * 64, wihWt + (size_t)l * 64 * 192, 192, 64);
    convert_transpose_kernel<<<cc(192 * 64), 256, 0, stream>>>(
        whh + (size_t)l * 192 * 64, whhWt + (size_t)l * 64 * 192, 192, 64);
  }

  // ---- layer 0 input: x = z_tables[0][z] ----
  embed0_kernel<<<cc((long long)NN * DD), 256, 0, stream>>>(z, tab_bf, xA);

  bf16* xcur = xA;
  bf16* xoth = xB;

  const int ROW_WAVES = NN / 16;                    // 31250
  const int TF_BLOCKS  = cc((long long)ROW_WAVES * 32);       // 16x64 tiles
  const int GRU_BLOCKS = cc((long long)ROW_WAVES * 2 * 32);   // 16x96 tiles

  for (int l = 0; l < NL; ++l) {
    if (l > 0) {
      tf_gemm_kernel<<<TF_BLOCKS, 256, 0, stream>>>(
          xcur, z, tab_bf + (size_t)l * 100 * DD,
          tfWt + (size_t)(l - 1) * 128 * 64, tf_b + (size_t)(l - 1) * 64, xoth);
      bf16* t = xcur; xcur = xoth; xoth = t;
    }
    // m = x @ W
    ggc_gemm_kernel<<<TF_BLOCKS, 256, 0, stream>>>(
        xcur, ggcWt + (size_t)l * 64 * 64, mbuf);
    // agg = segment_sum(m[src], dst)
    zero_kernel<<<cc((long long)NN * DD), 256, 0, stream>>>(agg, NN * DD);
    scatter_edges_kernel<<<cc((long long)EE * 32), 256, 0, stream>>>(esrc, edst, mbuf, agg);
    // gi = agg @ wih.T + bih
    gi_gemm_kernel<<<GRU_BLOCKS, 256, 0, stream>>>(
        agg, wihWt + (size_t)l * 64 * 192, bih + (size_t)l * 192, gi);
    // gh: cols 0..127 accumulate into gi, hn -> mbuf
    gh_gemm_kernel<<<GRU_BLOCKS, 256, 0, stream>>>(
        xcur, whhWt + (size_t)l * 64 * 192, bhh + (size_t)l * 192, gi, mbuf);
    // GRU elementwise update (in place)
    gru_kernel<<<cc((long long)NN * DD), 256, 0, stream>>>(xcur, gi, mbuf);
  }

  // ---- pooling + head ----
  zero_kernel<<<cc((long long)SS * DD), 256, 0, stream>>>(poolS, SS * DD);
  zero_kernel<<<cc((long long)GG * DD), 256, 0, stream>>>(poolG, GG * DD);
  pool_nodes_kernel<<<cc((long long)NN * 32), 256, 0, stream>>>(xcur, n2s, poolS);
  pool_sub_kernel<<<cc((long long)SS * 32), 256, 0, stream>>>(poolS, s2g, poolG);
  head_kernel<<<1, 64, 0, stream>>>(poolG, fc1W, fc1b, fc2W, fc2b, fc3W, fc3b, out);
}
